// Attentive_24223615549685
// MI455X (gfx1250) — compile-verified, matrix operand order checked by reading, not run on hardware
//
#include <hip/hip_runtime.h>
#include <hip/hip_bf16.h>
#include <math.h>
#include <stdint.h>

typedef __attribute__((ext_vector_type(16))) _Float16 v16h;
typedef __attribute__((ext_vector_type(8)))  _Float16 v8h;
typedef __attribute__((ext_vector_type(8)))  float    v8f;

#define BM 128
#define BN 128
#define BK 32
#define LDSS 40   // LDS row stride in halves (80B = 20 banks): 16B-aligned chunks,
                  // conflict-free ds_load_b128 fragment reads (m*20 mod 64 distinct)

// one async global->LDS 16-byte transfer (per lane), ASYNCcnt-tracked
#define ASYNC_LD_B128(ldsptr, goff_bytes, sbase)                              \
    asm volatile("global_load_async_to_lds_b128 %0, %1, %2"                   \
                 :: "v"((uint32_t)(uintptr_t)(ldsptr)),                       \
                    "v"((uint32_t)(goff_bytes)),                              \
                    "s"(sbase) : "memory")

#define WAIT_ASYNC() asm volatile("s_wait_asynccnt 0x0" ::: "memory")

// ---------------------------------------------------------------------------
// Kernel 1: Y[n, h*D+d] = (a[h,d]*x[n,d]) * rsqrt(max(sum (a*x)^2, eps)) / sqrt(H)
// One wave32 per (n, h) row; 8 waves per block.
// ---------------------------------------------------------------------------
__global__ __launch_bounds__(256) void normalize_kernel(
    const float* __restrict__ x, const float* __restrict__ av,
    _Float16* __restrict__ Y, int N, int D, int H, float inv_sqrt_H)
{
    int w    = blockIdx.x * 8 + (threadIdx.x >> 5);
    int lane = threadIdx.x & 31;
    if (w >= N * H) return;
    int n = w / H;
    int h = w % H;

    const float* xr = x  + (size_t)n * D;
    const float* ar = av + (size_t)h * D;

    float acc = 0.f;
    for (int j = lane; j < D; j += 32) {
        float v = ar[j] * xr[j];
        acc += v * v;
    }
    for (int off = 16; off > 0; off >>= 1)
        acc += __shfl_xor(acc, off, 32);

    float rs = rsqrtf(fmaxf(acc, 1e-12f)) * inv_sqrt_H;

    _Float16* yr = Y + (size_t)n * (D * H) + (size_t)h * D;
    for (int j = lane; j < D; j += 32) {
        float v = ar[j] * xr[j] * rs;
        yr[j] = (_Float16)v;
    }
}

// ---------------------------------------------------------------------------
// Kernel 2: C = Y * Y^T  (C: NxN f32, Y: NxK f16 row-major, K = H*D)
// 256 threads = 8 waves; 128x128 C tile per block; double-buffered LDS fed by
// async global->LDS DMA; v_wmma_f32_16x16x32_f16 with f32 accumulate.
// ---------------------------------------------------------------------------

// ISA 16-bit A/B fragment layout: lanes 0-15 hold K 0-7 & 16-23 of row lane;
// lanes 16-31 hold K 8-15 & 24-31 of row lane-16.
__device__ inline v16h load_frag(const _Float16* tile, int row, int sel)
{
    const _Float16* p = tile + row * LDSS;
    v8h lo = *(const v8h*)(p + sel * 8);        // K  0-7  or  8-15
    v8h hi = *(const v8h*)(p + 16 + sel * 8);   // K 16-23 or 24-31
    return __builtin_shufflevector(lo, hi, 0,1,2,3,4,5,6,7,8,9,10,11,12,13,14,15);
}

__global__ __launch_bounds__(256) void gemm_yyT(
    const _Float16* __restrict__ Y, float* __restrict__ C, int N, int K)
{
    __shared__ __align__(16) _Float16 As[2][BM * LDSS];
    __shared__ __align__(16) _Float16 Bs[2][BN * LDSS];

    const int tid    = threadIdx.x;
    const int lane   = tid & 31;
    const int wid    = tid >> 5;
    const int wave_m = wid >> 2;      // 0..1
    const int wave_n = wid & 3;       // 0..3
    const int sel    = lane >> 4;     // which K-half this lane holds
    const int lr     = lane & 15;

    const int rowA = blockIdx.x * BM;
    const int rowB = blockIdx.y * BN;

    // this thread's two 16B staging chunks: chunk c covers row c>>2, halves (c&3)*8
    const int r0 = tid >> 2,         cc0 = (tid & 3) * 8;
    const int r1 = (tid + 256) >> 2, cc1 = ((tid + 256) & 3) * 8;

    // issue one tile (A + B) of async DMA into buffer b for K-offset k0
    auto issue_tile = [&](int k0, int b) {
        uint32_t ga0 = (uint32_t)(((size_t)(rowA + r0) * K + k0 + cc0) * sizeof(_Float16));
        uint32_t ga1 = (uint32_t)(((size_t)(rowA + r1) * K + k0 + cc1) * sizeof(_Float16));
        uint32_t gb0 = (uint32_t)(((size_t)(rowB + r0) * K + k0 + cc0) * sizeof(_Float16));
        uint32_t gb1 = (uint32_t)(((size_t)(rowB + r1) * K + k0 + cc1) * sizeof(_Float16));
        ASYNC_LD_B128(&As[b][r0 * LDSS + cc0], ga0, Y);
        ASYNC_LD_B128(&As[b][r1 * LDSS + cc1], ga1, Y);
        ASYNC_LD_B128(&Bs[b][r0 * LDSS + cc0], gb0, Y);
        ASYNC_LD_B128(&Bs[b][r1 * LDSS + cc1], gb1, Y);
    };

    v8f acc[4][2] = {};

    issue_tile(0, 0);
    WAIT_ASYNC();
    __syncthreads();

    int buf = 0;
    for (int k0 = 0; k0 < K; k0 += BK) {
        const bool more = (k0 + BK) < K;
        // kick off next tile's DMA; overlaps fragment loads + WMMAs below
        if (more) issue_tile(k0 + BK, buf ^ 1);

        v16h af[4], bf[2];
        #pragma unroll
        for (int sm = 0; sm < 4; ++sm)
            af[sm] = load_frag(As[buf], wave_m * 64 + sm * 16 + lr, sel);
        #pragma unroll
        for (int sn = 0; sn < 2; ++sn)
            bf[sn] = load_frag(Bs[buf], wave_n * 32 + sn * 16 + lr, sel);

        #pragma unroll
        for (int sm = 0; sm < 4; ++sm)
            #pragma unroll
            for (int sn = 0; sn < 2; ++sn)
                acc[sm][sn] = __builtin_amdgcn_wmma_f32_16x16x32_f16(
                    false, af[sm], false, bf[sn],
                    (short)0, acc[sm][sn], false, false);

        if (more) {
            WAIT_ASYNC();       // next tile landed in LDS
            __syncthreads();    // visible to all waves; prev reads done
            buf ^= 1;
        }
    }

    // D layout: 8 VGPRs; lanes 0-15 -> M=i, lanes 16-31 -> M=8+i; N = lane&15
    #pragma unroll
    for (int sm = 0; sm < 4; ++sm) {
        #pragma unroll
        for (int sn = 0; sn < 2; ++sn) {
            int row0 = rowA + wave_m * 64 + sm * 16 + sel * 8;
            int col  = rowB + wave_n * 32 + sn * 16 + lr;
            #pragma unroll
            for (int i = 0; i < 8; ++i) {
                __builtin_nontemporal_store(
                    acc[sm][sn][i], &C[(size_t)(row0 + i) * N + col]);
            }
        }
    }
}

// ---------------------------------------------------------------------------
extern "C" void kernel_launch(void* const* d_in, const int* in_sizes, int n_in,
                              void* d_out, int out_size, void* d_ws, size_t ws_size,
                              hipStream_t stream) {
    const float* x  = (const float*)d_in[0];   // [N, D]
    const float* av = (const float*)d_in[1];   // [H, D]
    float* C        = (float*)d_out;           // [N, N]
    _Float16* Y     = (_Float16*)d_ws;         // [N, H*D] f16 staging (16 MB)

    int N  = (int)llround(sqrt((double)out_size));   // 8192
    int D  = in_sizes[0] / N;                        // 256
    int H  = in_sizes[1] / D;                        // 4
    int K  = H * D;                                  // 1024
    float inv_sqrt_H = rsqrtf((float)H);

    int waves  = N * H;
    int blocks = (waves + 7) / 8;
    normalize_kernel<<<blocks, 256, 0, stream>>>(x, av, Y, N, D, H, inv_sqrt_H);

    dim3 grid(N / BM, N / BN);
    gemm_yyT<<<grid, 256, 0, stream>>>(Y, C, N, K);
}
